// Backbone_19963007992433
// MI455X (gfx1250) — 8-bit matrix operand order repaired and checked by reading, compiled, not run on hardware
//
#include <hip/hip_runtime.h>
#include <stdint.h>


typedef int   v16i __attribute__((ext_vector_type(16)));
typedef float v8f  __attribute__((ext_vector_type(8)));

#define TN_NODE 512
#define SEQ     1024
#define NB      8
#define KBYTES  512
#define FP8_ONE 0x38
#define TM      128
#define TN      64
#define LROW    144
#define KC      128
#define NCHUNK  (KBYTES / KC)

#define OFF_AT  0
#define OFF_AL  18432
#define OFF_BT  36864
#define OFF_BL  46080
#define SM_BYTES 55296

__global__ __launch_bounds__(128)
void build_pos_kernel(const int* __restrict__ zt, const int* __restrict__ zl,
                      const int* __restrict__ ind,
                      uint8_t* __restrict__ posT, uint8_t* __restrict__ posL,
                      float* __restrict__ sT, float* __restrict__ sL,
                      float* __restrict__ padF)
{
    const int row = blockIdx.x;
    uint8_t* rT = posT + (size_t)row * KBYTES;
    uint8_t* rL = posL + (size_t)row * KBYTES;
    ((uint32_t*)rT)[threadIdx.x] = 0u;
    ((uint32_t*)rL)[threadIdx.x] = 0u;
    __syncthreads();
    if (threadIdx.x == 0) {
        int ids[4]; int s = 0;
        #pragma unroll
        for (int d = 0; d < 4; ++d) {
            int id = zt[row * 4 + d]; ids[d] = id;
            if (id >= 0 && id < TN_NODE) {
                bool dup = false;
                for (int e = 0; e < d; ++e) dup |= (ids[e] == id);
                if (!dup) { rT[id] = (uint8_t)FP8_ONE; ++s; }
            }
        }
        sT[row] = (float)s;
    } else if (threadIdx.x == 1) {
        int ids[4]; int s = 0;
        #pragma unroll
        for (int d = 0; d < 4; ++d) {
            int id = zl[row * 4 + d]; ids[d] = id;
            if (id >= 0 && id < TN_NODE) {
                bool dup = false;
                for (int e = 0; e < d; ++e) dup |= (ids[e] == id);
                if (!dup) { rL[id] = (uint8_t)FP8_ONE; ++s; }
            }
        }
        sL[row] = (float)s;
    } else if (threadIdx.x == 2) {
        padF[row] = (ind[row] == 0) ? 1.0f : 0.0f;
    }
}

static __device__ __forceinline__ v16i loadA128(const uint8_t* p)
{
    v16i r;
    #pragma unroll
    for (int h = 0; h < 2; ++h)
        #pragma unroll
        for (int q = 0; q < 4; ++q) {
            uint2 d = *(const uint2*)(p + h * 64 + 16 * q);
            r[h * 8 + 2 * q]     = (int)d.x;
            r[h * 8 + 2 * q + 1] = (int)d.y;
        }
    return r;
}
static __device__ __forceinline__ v16i loadB128(const uint8_t* p)
{
    v16i r;
    #pragma unroll
    for (int s = 0; s < 4; ++s) {
        uint4 d = *(const uint4*)(p + 32 * s);
        r[4 * s]     = (int)d.x;
        r[4 * s + 1] = (int)d.y;
        r[4 * s + 2] = (int)d.z;
        r[4 * s + 3] = (int)d.w;
    }
    return r;
}

static __device__ __forceinline__ void stage16(const uint8_t* __restrict__ g,
                                               uint8_t* sm, int idx, int koff)
{
    const int r = idx >> 3, q = idx & 7;
    uint4 v = *(const uint4*)(g + (size_t)r * KBYTES + koff + q * 16);
    *(uint4*)(sm + r * LROW + q * 16) = v;
}

__global__ __launch_bounds__(256)
void attn_wmma_kernel(const uint8_t* __restrict__ posT, const uint8_t* __restrict__ posL,
                      const float* __restrict__ sT, const float* __restrict__ sL,
                      const float* __restrict__ padF, const float* __restrict__ pdist,
                      float* __restrict__ out)
{
    const int bN  = blockIdx.x * TN;
    const int bM  = blockIdx.y * TM;
    const int bb  = blockIdx.z;
    const int tid = threadIdx.x;
    const int lane = tid & 31;
    const int w    = tid >> 5;
    const int wr   = w >> 1;
    const int wc   = w & 1;

    __shared__ __align__(16) uint8_t smem[SM_BYTES];
    __shared__ float sMv[3][TM];
    __shared__ float sNv[3][TN];

    const size_t rowBaseM = (size_t)(bb * SEQ + bM);
    const size_t rowBaseN = (size_t)(bb * SEQ + bN);

    for (int i = tid; i < TM; i += 256) {
        sMv[0][i] = sT[rowBaseM + i];
        sMv[1][i] = sL[rowBaseM + i];
        sMv[2][i] = padF[rowBaseM + i];
    }
    for (int i = tid; i < TN; i += 256) {
        sNv[0][i] = sT[rowBaseN + i];
        sNv[1][i] = sL[rowBaseN + i];
        sNv[2][i] = padF[rowBaseN + i];
    }

    const v8f zacc = {0.f, 0.f, 0.f, 0.f, 0.f, 0.f, 0.f, 0.f};
    v8f accT[2][2], accL[2][2];
    #pragma unroll
    for (int mi = 0; mi < 2; ++mi)
        #pragma unroll
        for (int ni = 0; ni < 2; ++ni) { accT[mi][ni] = zacc; accL[mi][ni] = zacc; }

    const int hb8  = (lane >> 4) * 8;
    const int hb16 = (lane >> 4) * 16;
    const int l15  = lane & 15;

    const uint8_t* gAT = posT + rowBaseM * KBYTES;
    const uint8_t* gAL = posL + rowBaseM * KBYTES;
    const uint8_t* gBT = posT + rowBaseN * KBYTES;
    const uint8_t* gBL = posL + rowBaseN * KBYTES;

    for (int c = 0; c < NCHUNK; ++c) {
        __syncthreads();
        const int koff = c * KC;
        #pragma unroll
        for (int k = 0; k < 4; ++k) {
            stage16(gAT, smem + OFF_AT, tid + 256 * k, koff);
            stage16(gAL, smem + OFF_AL, tid + 256 * k, koff);
        }
        #pragma unroll
        for (int k = 0; k < 2; ++k) {
            stage16(gBT, smem + OFF_BT, tid + 256 * k, koff);
            stage16(gBL, smem + OFF_BL, tid + 256 * k, koff);
        }
        __syncthreads();

        {
            v16i a[2], b[2];
            #pragma unroll
            for (int mi = 0; mi < 2; ++mi)
                a[mi] = loadB128(smem + OFF_AT + (wr * 32 + mi * 16 + l15) * LROW + hb16);
            #pragma unroll
            for (int ni = 0; ni < 2; ++ni)
                b[ni] = loadB128(smem + OFF_BT + (wc * 32 + ni * 16 + l15) * LROW + hb16);
            #pragma unroll
            for (int mi = 0; mi < 2; ++mi)
                #pragma unroll
                for (int ni = 0; ni < 2; ++ni)
                    accT[mi][ni] = __builtin_amdgcn_wmma_f32_16x16x128_fp8_fp8(
                        a[mi], b[ni], (short)0, accT[mi][ni], false, false);
        }
        {
            v16i a[2], b[2];
            #pragma unroll
            for (int mi = 0; mi < 2; ++mi)
                a[mi] = loadB128(smem + OFF_AL + (wr * 32 + mi * 16 + l15) * LROW + hb16);
            #pragma unroll
            for (int ni = 0; ni < 2; ++ni)
                b[ni] = loadB128(smem + OFF_BL + (wc * 32 + ni * 16 + l15) * LROW + hb16);
            #pragma unroll
            for (int mi = 0; mi < 2; ++mi)
                #pragma unroll
                for (int ni = 0; ni < 2; ++ni)
                    accL[mi][ni] = __builtin_amdgcn_wmma_f32_16x16x128_fp8_fp8(
                        a[mi], b[ni], (short)0, accL[mi][ni], false, false);
        }
    }

    const float dist = *pdist;
    float* outB = out + (size_t)bb * SEQ * SEQ;
    const int hiM = (lane >> 4) * 8;
    #pragma unroll
    for (int mi = 0; mi < 2; ++mi)
        #pragma unroll
        for (int ni = 0; ni < 2; ++ni)
            #pragma unroll
            for (int v = 0; v < 8; ++v) {
                const int mloc = wr * 32 + mi * 16 + v + hiM;
                const int nloc = wc * 32 + ni * 16 + l15;
                const float vt = sMv[0][mloc] + sNv[0][nloc] - 2.0f * accT[mi][ni][v];
                const float vl = sMv[1][mloc] + sNv[1][nloc] - 2.0f * accL[mi][ni][v];
                const float pm = fmaxf(sMv[2][mloc], sNv[2][nloc]) * dist;
                outB[(size_t)(bM + mloc) * SEQ + (bN + nloc)] = fmaxf(vt, vl) + pm;
            }
}

extern "C" void kernel_launch(void* const* d_in, const int* in_sizes, int n_in,
                              void* d_out, int out_size, void* d_ws, size_t ws_size,
                              hipStream_t stream)
{
    (void)in_sizes; (void)n_in; (void)out_size; (void)ws_size;
    const int*   zt    = (const int*)d_in[0];
    const int*   zl    = (const int*)d_in[1];
    const int*   ind   = (const int*)d_in[2];
    const float* pdist = (const float*)d_in[3];
    float*       out   = (float*)d_out;

    uint8_t* posT = (uint8_t*)d_ws;
    uint8_t* posL = posT + (size_t)NB * SEQ * KBYTES;
    float*   sT   = (float*)(posL + (size_t)NB * SEQ * KBYTES);
    float*   sL   = sT + NB * SEQ;
    float*   padF = sL + NB * SEQ;

    hipLaunchKernelGGL(build_pos_kernel, dim3(NB * SEQ), dim3(128), 0, stream,
                       zt, zl, ind, posT, posL, sT, sL, padF);
    hipLaunchKernelGGL(attn_wmma_kernel, dim3(SEQ / TN, SEQ / TM, NB), dim3(256), 0, stream,
                       posT, posL, sT, sL, padF, pdist, out);
}
